// DenoiseLSTM_8589935063
// MI455X (gfx1250) — matrix operand order repaired and checked by reading, not compiled, not run on hardware
//
#include <hip/hip_runtime.h>

typedef __attribute__((ext_vector_type(16))) _Float16 v16h;
typedef __attribute__((ext_vector_type(8)))  float    v8f;
typedef _Float16 half_t;

#define D_EMBC 128
#define D_ENCC 256
#define D_DECC 512
#define VOCABC 32000
#define BBC 32
#define SSC 128
#define TTC 128

// ---------------------------------------------------------------------------
// WMMA fragment helpers (wave32; A/B use contiguous k = 16*(lane>>4) + e per lane half).
// Base-pointer style: per-lane bases hoisted out of K loops so per-iteration
// addressing folds into load immediate offsets.
// ---------------------------------------------------------------------------

// A fragment base: row-major A (stride lda), 16 rows at m0.
// lane<16: elems 0..15 -> K=k0+0..15
// lane>=16: elems 0..15 -> K=k0+16..31
__device__ __forceinline__ const half_t* a_frag_base(const half_t* A, int lda, int m0) {
    const int lane = threadIdx.x & 31;
    return A + (size_t)(m0 + (lane & 15)) * lda + (lane >> 4) * 16;
}
__device__ __forceinline__ v16h load_a_at(const half_t* base, int k0) {
    v16h a;
#pragma unroll
    for (int e = 0; e < 16; ++e) a[e] = base[k0 + e];
    return a;
}

// B fragment representing W^T: B[k][n] = W[n][k], W row-major [N][ldw].
// lane<16: col=n0+lane, K=k0..k0+15 ; lane>=16: col=n0+lane-16, K=k0+16..31
__device__ __forceinline__ const half_t* b_frag_base(const half_t* W, int ldw, int n0) {
    const int lane = threadIdx.x & 31;
    return W + (size_t)(n0 + (lane & 15)) * ldw + (lane >> 4) * 16;
}
__device__ __forceinline__ v16h load_b_at(const half_t* base, int k0) {
    v16h b;
#pragma unroll
    for (int e = 0; e < 16; ++e) b[e] = base[k0 + e];
    return b;
}

// C/D fragment: elem v -> row m0 + 8*(lane>>4) + v, col n0 + (lane&15)
__device__ __forceinline__ v8f load_c_frag(const float* C, int ldc, int m0, int n0) {
    const int lane = threadIdx.x & 31;
    const int col  = n0 + (lane & 15);
    const int rb   = m0 + ((lane >> 4) << 3);
    v8f c;
#pragma unroll
    for (int v = 0; v < 8; ++v) c[v] = C[(rb + v) * ldc + col];
    return c;
}

__device__ __forceinline__ v8f wmma16(v16h a, v16h b, v8f c) {
    return __builtin_amdgcn_wmma_f32_16x16x32_f16(false, a, false, b, (short)0, c, false, false);
}

__device__ __forceinline__ float sigf(float x)   { return 1.0f / (1.0f + __expf(-x)); }
__device__ __forceinline__ float lreluf(float x) { return x >= 0.0f ? x : 0.1f * x; }

// ---------------------------------------------------------------------------
// Utility kernels
// ---------------------------------------------------------------------------

__global__ void f32_to_f16_kernel(const float* __restrict__ src, half_t* __restrict__ dst, int n) {
    int i = blockIdx.x * blockDim.x + threadIdx.x;
    if (i < n) dst[i] = (half_t)src[i];
}

// embE[(s*32+b)*128 + e] = tok_emb[inp[b][s]][e]
__global__ void gather_enc_emb_kernel(const int* __restrict__ inp, const float* __restrict__ tok_emb,
                                      half_t* __restrict__ embE) {
    int i = blockIdx.x * blockDim.x + threadIdx.x;
    if (i >= SSC * BBC * D_EMBC) return;
    int e = i & 127, sb = i >> 7;
    int b = sb & 31, s = sb >> 5;
    int tok = inp[b * SSC + s];
    embE[i] = (half_t)tok_emb[(size_t)tok * D_EMBC + e];
}

// embD[(t*32+b)*128 + e] = (t==0) ? start_emb[e] : tok_emb[x[b][t-1]][e]
__global__ void gather_dec_emb_kernel(const int* __restrict__ x, const float* __restrict__ tok_emb,
                                      const float* __restrict__ start_emb, half_t* __restrict__ embD) {
    int i = blockIdx.x * blockDim.x + threadIdx.x;
    if (i >= TTC * BBC * D_EMBC) return;
    int e = i & 127, tb = i >> 7;
    int b = tb & 31, t = tb >> 5;
    float v;
    if (t == 0) v = start_emb[e];
    else        v = tok_emb[(size_t)x[b * TTC + (t - 1)] * D_EMBC + e];
    embD[i] = (half_t)v;
}

// ---------------------------------------------------------------------------
// Generic WMMA GEMM: out[M][N] = act(A[M][K] @ W[N][K]^T + bias1 + bias2)
// One wave per 16x64 strip (4 column tiles, A reuse). K loop unrolled by 2
// as a ping-pong over two named fragment sets: stage A computes from set 0
// while loading set 1 and vice versa -> no rotation copies, loads for k+32
// always in flight under the WMMAs for k. Requires K % 64 == 0 (all uses).
// ---------------------------------------------------------------------------
__global__ void wmma_gemm_kernel(const half_t* __restrict__ A, int lda,
                                 const half_t* __restrict__ W, int ldw,
                                 const float* __restrict__ bias1,
                                 const float* __restrict__ bias2,
                                 float* __restrict__ outF,
                                 half_t* __restrict__ outH, int ldo,
                                 int M, int N, int K, int act) {
    const int lane   = threadIdx.x & 31;
    const int wid    = blockIdx.x * (blockDim.x >> 5) + (threadIdx.x >> 5);
    const int strips = N >> 6;
    if (wid >= (M >> 4) * strips) return;
    const int m0 = (wid / strips) << 4;
    const int n0 = (wid % strips) << 6;

    const half_t* aB = a_frag_base(A, lda, m0);
    const half_t* bB[4];
#pragma unroll
    for (int j = 0; j < 4; ++j) bB[j] = b_frag_base(W, ldw, n0 + j * 16);

    v8f acc[4] = {};
    v16h a0 = load_a_at(aB, 0), a1;
    v16h b0[4], b1[4];
#pragma unroll
    for (int j = 0; j < 4; ++j) b0[j] = load_b_at(bB[j], 0);

    for (int k0 = 0; k0 < K; k0 += 64) {
        const int kA = k0 + 32;
        a1 = load_a_at(aB, kA);
#pragma unroll
        for (int j = 0; j < 4; ++j) b1[j] = load_b_at(bB[j], kA);
#pragma unroll
        for (int j = 0; j < 4; ++j) acc[j] = wmma16(a0, b0[j], acc[j]);
        const int kB = (k0 + 64 < K) ? (k0 + 64) : 0;   // last iter: harmless refetch
        a0 = load_a_at(aB, kB);
#pragma unroll
        for (int j = 0; j < 4; ++j) b0[j] = load_b_at(bB[j], kB);
#pragma unroll
        for (int j = 0; j < 4; ++j) acc[j] = wmma16(a1, b1[j], acc[j]);
    }

    const int rb = m0 + ((lane >> 4) << 3);
#pragma unroll
    for (int j = 0; j < 4; ++j) {
        const int col = n0 + j * 16 + (lane & 15);
        float badd = (bias1 ? bias1[col] : 0.0f) + (bias2 ? bias2[col] : 0.0f);
#pragma unroll
        for (int v = 0; v < 8; ++v) {
            float xv = acc[j][v] + badd;
            if (act) xv = lreluf(xv);
            if (outF) outF[(size_t)(rb + v) * ldo + col] = xv;
            else      outH[(size_t)(rb + v) * ldo + col] = (half_t)xv;
        }
    }
}

// ---------------------------------------------------------------------------
// Vocab GEMM: logits = hid[4096][512] @ W2[32000][512]^T, rows r=t*32+b
// remapped to d_out[b][t][v]. K=512 fully unrolled, ping-pong buffered.
// ---------------------------------------------------------------------------
__global__ void vocab_gemm_kernel(const half_t* __restrict__ hid,
                                  const half_t* __restrict__ W2,
                                  float* __restrict__ out) {
    const int lane   = threadIdx.x & 31;
    const int wid    = blockIdx.x * (blockDim.x >> 5) + (threadIdx.x >> 5);
    const int strips = VOCABC >> 6;   // 500
    if (wid >= ((TTC * BBC) >> 4) * strips) return;
    const int m0 = (wid / strips) << 4;
    const int n0 = (wid % strips) << 6;

    const half_t* aB = a_frag_base(hid, D_DECC, m0);
    const half_t* bB[4];
#pragma unroll
    for (int j = 0; j < 4; ++j) bB[j] = b_frag_base(W2, D_DECC, n0 + j * 16);

    v8f acc[4] = {};
    v16h a0 = load_a_at(aB, 0), a1;
    v16h b0[4], b1[4];
#pragma unroll
    for (int j = 0; j < 4; ++j) b0[j] = load_b_at(bB[j], 0);

#pragma unroll
    for (int kp = 0; kp < 8; ++kp) {              // 8 x 64 = K=512
        const int kA = kp * 64 + 32;
        a1 = load_a_at(aB, kA);
#pragma unroll
        for (int j = 0; j < 4; ++j) b1[j] = load_b_at(bB[j], kA);
#pragma unroll
        for (int j = 0; j < 4; ++j) acc[j] = wmma16(a0, b0[j], acc[j]);
        const int kB = (kp < 7) ? (kp + 1) * 64 : 0;
        a0 = load_a_at(aB, kB);
#pragma unroll
        for (int j = 0; j < 4; ++j) b0[j] = load_b_at(bB[j], kB);
#pragma unroll
        for (int j = 0; j < 4; ++j) acc[j] = wmma16(a1, b1[j], acc[j]);
    }

    const int rb = m0 + ((lane >> 4) << 3);
#pragma unroll
    for (int j = 0; j < 4; ++j) {
        const int col = n0 + j * 16 + (lane & 15);
#pragma unroll
        for (int v = 0; v < 8; ++v) {
            int r = rb + v;
            int b = r & 31, t = r >> 5;
            out[((size_t)b * TTC + t) * VOCABC + col] = acc[j][v];
        }
    }
}

// ---------------------------------------------------------------------------
// Encoder LSTM (persistent, one block per direction, 512 thr = 16 waves).
// Each wave owns 2 groups = (m0,n0) offsets whose 4 gate-quarter tiles (i/f/g/o)
// stay in registers; activations applied on WMMA fragments; c-state in VGPRs.
// Whh fragment loads ping-pong buffered across the K loop (8 chunks -> 4 iters).
// ---------------------------------------------------------------------------
__global__ void enc_rnn_kernel(const float* __restrict__ Xe,    // [2][S][B][1024] (x@Wih^T + biases)
                               const half_t* __restrict__ Whh,  // [2][1024][256]
                               const int* __restrict__ label_i,
                               const float* __restrict__ enc_style_emb, // [NC][512]
                               float* __restrict__ memory,      // [B][S][512]
                               float* __restrict__ c_out) {     // [2][B][256]
    const int dir  = blockIdx.x;
    const int tid  = threadIdx.x;     // 512
    const int wave = tid >> 5;
    const int lane = tid & 31;
    __shared__ half_t hf[BBC * D_ENCC];

    const float*  X = Xe  + (size_t)dir * SSC * BBC * 1024;
    const half_t* W = Whh + (size_t)dir * 1024 * D_ENCC;

    for (int i = tid; i < BBC * D_ENCC; i += 512) {
        int b = i >> 8, j = i & 255;
        hf[i] = (half_t)enc_style_emb[(size_t)label_i[b] * (2 * D_ENCC) + dir * D_ENCC + j];
    }
    // per-group bases (per-lane, hoisted out of the step loop)
    const half_t* aB[2];
    const half_t* bB[2][4];
#pragma unroll
    for (int gidx = 0; gidx < 2; ++gidx) {
        const int g  = wave + gidx * 16;
        const int m0 = (g & 1) * 16;
        const int n0 = (g >> 1) * 16;
        aB[gidx] = a_frag_base(hf, D_ENCC, m0);
#pragma unroll
        for (int q = 0; q < 4; ++q) bB[gidx][q] = b_frag_base(W, D_ENCC, q * D_ENCC + n0);
    }
    float cst[2][8];
#pragma unroll
    for (int g = 0; g < 2; ++g)
#pragma unroll
        for (int v = 0; v < 8; ++v) cst[g][v] = 0.0f;
    __syncthreads();

    for (int s = 0; s < SSC; ++s) {
        const int sIn = dir ? (SSC - 1 - s) : s;
        const float* Xs = X + (size_t)sIn * BBC * 1024;
        float hloc[2][8];
#pragma unroll
        for (int gidx = 0; gidx < 2; ++gidx) {
            const int g  = wave + gidx * 16;
            const int m0 = (g & 1) * 16;
            const int n0 = (g >> 1) * 16;
            v8f acc[4];
#pragma unroll
            for (int q = 0; q < 4; ++q) acc[q] = load_c_frag(Xs, 1024, m0, q * D_ENCC + n0);
            v16h a0 = load_a_at(aB[gidx], 0), a1;
            v16h b0[4], b1[4];
#pragma unroll
            for (int q = 0; q < 4; ++q) b0[q] = load_b_at(bB[gidx][q], 0);
#pragma unroll
            for (int kp = 0; kp < 4; ++kp) {          // 4 x 64 = K=256
                const int kA = kp * 64 + 32;
                a1 = load_a_at(aB[gidx], kA);
#pragma unroll
                for (int q = 0; q < 4; ++q) b1[q] = load_b_at(bB[gidx][q], kA);
#pragma unroll
                for (int q = 0; q < 4; ++q) acc[q] = wmma16(a0, b0[q], acc[q]);
                const int kB = (kp < 3) ? (kp + 1) * 64 : 0;
                a0 = load_a_at(aB[gidx], kB);
#pragma unroll
                for (int q = 0; q < 4; ++q) b0[q] = load_b_at(bB[gidx][q], kB);
#pragma unroll
                for (int q = 0; q < 4; ++q) acc[q] = wmma16(a1, b1[q], acc[q]);
            }
#pragma unroll
            for (int v = 0; v < 8; ++v) {
                float gI = acc[0][v], gF = acc[1][v], gG = acc[2][v], gO = acc[3][v];
                float c  = sigf(gF) * cst[gidx][v] + sigf(gI) * tanhf(gG);
                cst[gidx][v]  = c;
                hloc[gidx][v] = sigf(gO) * tanhf(c);
            }
        }
        __syncthreads();  // everyone done reading old hf
#pragma unroll
        for (int gidx = 0; gidx < 2; ++gidx) {
            const int g   = wave + gidx * 16;
            const int m0  = (g & 1) * 16;
            const int n0  = (g >> 1) * 16;
            const int rb  = m0 + ((lane >> 4) << 3);
            const int col = n0 + (lane & 15);
#pragma unroll
            for (int v = 0; v < 8; ++v) {
                float h = hloc[gidx][v];
                hf[(rb + v) * D_ENCC + col] = (half_t)h;
                memory[((size_t)(rb + v) * SSC + sIn) * (2 * D_ENCC) + dir * D_ENCC + col] = h;
            }
        }
        __syncthreads();
    }
#pragma unroll
    for (int gidx = 0; gidx < 2; ++gidx) {
        const int g   = wave + gidx * 16;
        const int m0  = (g & 1) * 16;
        const int n0  = (g >> 1) * 16;
        const int rb  = m0 + ((lane >> 4) << 3);
        const int col = n0 + (lane & 15);
#pragma unroll
        for (int v = 0; v < 8; ++v)
            c_out[((size_t)dir * BBC + (rb + v)) * D_ENCC + col] = cst[gidx][v];
    }
}

// ---------------------------------------------------------------------------
// c_t = lrelu([cf|cb] @ W_tr^T), h_t = style_emb[label]. 256 thr = 8 waves.
// ---------------------------------------------------------------------------
__global__ void ctran_kernel(const float* __restrict__ c_out,  // [2][B][256]
                             const half_t* __restrict__ Wtr,   // [512][512]
                             const int* __restrict__ label,
                             const float* __restrict__ style_emb,
                             float* __restrict__ cd,            // [B][512]
                             float* __restrict__ hd) {          // [B][512]
    const int tid  = threadIdx.x;
    const int wave = tid >> 5;
    const int lane = tid & 31;
    __shared__ half_t cc[BBC * D_DECC];
    for (int i = tid; i < BBC * D_DECC; i += 256) {
        int b = i >> 9, k = i & 511;
        float v = (k < 256) ? c_out[b * 256 + k] : c_out[BBC * 256 + b * 256 + (k - 256)];
        cc[i] = (half_t)v;
        hd[i] = style_emb[(size_t)label[b] * D_DECC + k];
    }
    __syncthreads();
    for (int it = 0; it < 8; ++it) {
        int t  = wave * 8 + it;
        int m0 = (t & 1) * 16;
        int n0 = (t >> 1) * 16;
        const half_t* aB = a_frag_base(cc, D_DECC, m0);
        const half_t* bB = b_frag_base(Wtr, D_DECC, n0);
        v8f acc = {};
        v16h a0 = load_a_at(aB, 0), a1;
        v16h b0 = load_b_at(bB, 0), b1;
#pragma unroll
        for (int kp = 0; kp < 8; ++kp) {              // 8 x 64 = K=512
            const int kA = kp * 64 + 32;
            a1 = load_a_at(aB, kA);
            b1 = load_b_at(bB, kA);
            acc = wmma16(a0, b0, acc);
            const int kB = (kp < 7) ? (kp + 1) * 64 : 0;
            a0 = load_a_at(aB, kB);
            b0 = load_b_at(bB, kB);
            acc = wmma16(a1, b1, acc);
        }
        int rb = m0 + ((lane >> 4) << 3), col = n0 + (lane & 15);
#pragma unroll
        for (int v = 0; v < 8; ++v)
            cd[(rb + v) * D_DECC + col] = lreluf(acc[v]);
    }
}

// ---------------------------------------------------------------------------
// Decoder LSTM + attention (persistent single block, 1024 thr = 32 waves).
// Gates [32][2048] fragment-resident, Whh loads ping-pong buffered (16 chunks
// -> 8 iters); attention in VALU with contiguous access; emits ffn=[h|ctx] f16.
// ---------------------------------------------------------------------------
__global__ void dec_rnn_kernel(const float* __restrict__ Xd,    // [T][B][2048]
                               const half_t* __restrict__ Whh,  // [2048][512]
                               const float* __restrict__ hd,
                               const float* __restrict__ cd,
                               const float* __restrict__ memory, // [B][S][512]
                               half_t* __restrict__ ffn) {       // [T*B][1024]
    const int tid  = threadIdx.x;   // 1024
    const int wave = tid >> 5;
    const int lane = tid & 31;
    __shared__ half_t hf[BBC * D_DECC];   // 32 KB
    __shared__ float  att[BBC * SSC];     // 16 KB

    for (int i = tid; i < BBC * D_DECC; i += 1024) hf[i] = (half_t)hd[i];

    const half_t* aB[2];
    const half_t* bB[2][4];
    float cst[2][8];
#pragma unroll
    for (int gidx = 0; gidx < 2; ++gidx) {
        const int g   = wave + gidx * 32;
        const int m0  = (g & 1) * 16;
        const int n0  = (g >> 1) * 16;
        aB[gidx] = a_frag_base(hf, D_DECC, m0);
#pragma unroll
        for (int q = 0; q < 4; ++q) bB[gidx][q] = b_frag_base(Whh, D_DECC, q * D_DECC + n0);
        const int rb  = m0 + ((lane >> 4) << 3);
        const int col = n0 + (lane & 15);
#pragma unroll
        for (int v = 0; v < 8; ++v) cst[gidx][v] = cd[(rb + v) * D_DECC + col];
    }
    __syncthreads();

    const float scale = 0.044194173824159216f;  // 1/sqrt(512)
    for (int t = 0; t < TTC; ++t) {
        const float* Xt = Xd + (size_t)t * BBC * 2048;
        float hloc[2][8];
#pragma unroll
        for (int gidx = 0; gidx < 2; ++gidx) {
            const int g  = wave + gidx * 32;
            const int m0 = (g & 1) * 16;
            const int n0 = (g >> 1) * 16;
            v8f acc[4];
#pragma unroll
            for (int q = 0; q < 4; ++q) acc[q] = load_c_frag(Xt, 2048, m0, q * D_DECC + n0);
            v16h a0 = load_a_at(aB[gidx], 0), a1;
            v16h b0[4], b1[4];
#pragma unroll
            for (int q = 0; q < 4; ++q) b0[q] = load_b_at(bB[gidx][q], 0);
#pragma unroll
            for (int kp = 0; kp < 8; ++kp) {          // 8 x 64 = K=512
                const int kA = kp * 64 + 32;
                a1 = load_a_at(aB[gidx], kA);
#pragma unroll
                for (int q = 0; q < 4; ++q) b1[q] = load_b_at(bB[gidx][q], kA);
#pragma unroll
                for (int q = 0; q < 4; ++q) acc[q] = wmma16(a0, b0[q], acc[q]);
                const int kB = (kp < 7) ? (kp + 1) * 64 : 0;
                a0 = load_a_at(aB[gidx], kB);
#pragma unroll
                for (int q = 0; q < 4; ++q) b0[q] = load_b_at(bB[gidx][q], kB);
#pragma unroll
                for (int q = 0; q < 4; ++q) acc[q] = wmma16(a1, b1[q], acc[q]);
            }
#pragma unroll
            for (int v = 0; v < 8; ++v) {
                float gI = acc[0][v], gF = acc[1][v], gG = acc[2][v], gO = acc[3][v];
                float c  = sigf(gF) * cst[gidx][v] + sigf(gI) * tanhf(gG);
                cst[gidx][v]  = c;
                hloc[gidx][v] = sigf(gO) * tanhf(c);
            }
        }
        __syncthreads();
#pragma unroll
        for (int gidx = 0; gidx < 2; ++gidx) {
            const int g   = wave + gidx * 32;
            const int m0  = (g & 1) * 16;
            const int n0  = (g >> 1) * 16;
            const int rb  = m0 + ((lane >> 4) << 3);
            const int col = n0 + (lane & 15);
#pragma unroll
            for (int v = 0; v < 8; ++v) {
                float h = hloc[gidx][v];
                hf[(rb + v) * D_DECC + col] = (half_t)h;
                ffn[((size_t)t * BBC + (rb + v)) * 1024 + col] = (half_t)h;
            }
        }
        __syncthreads();
        // attention scores: thread -> (b = wave, s0 = lane*4): h reused 4x per LDS read
        {
            const int b  = wave;
            const int s0 = lane * 4;
            const float*  mrow = memory + ((size_t)b * SSC + s0) * D_DECC;
            const half_t* hb   = hf + b * D_DECC;
            float a4[4] = {0.0f, 0.0f, 0.0f, 0.0f};
            for (int j = 0; j < D_DECC; ++j) {
                float hj = (float)hb[j];
#pragma unroll
                for (int k = 0; k < 4; ++k) a4[k] += hj * mrow[(size_t)k * D_DECC + j];
            }
#pragma unroll
            for (int k = 0; k < 4; ++k) att[b * SSC + s0 + k] = a4[k] * scale;
        }
        __syncthreads();
        // softmax: wave b handles row b (32 waves, 32 rows, 4 elems/lane)
        {
            const int b = wave;
            float e[4];
            float m = -1e30f;
#pragma unroll
            for (int k = 0; k < 4; ++k) m = fmaxf(m, att[b * SSC + lane + k * 32]);
#pragma unroll
            for (int off = 16; off; off >>= 1) m = fmaxf(m, __shfl_xor(m, off));
            float sum = 0.0f;
#pragma unroll
            for (int k = 0; k < 4; ++k) { e[k] = __expf(att[b * SSC + lane + k * 32] - m); sum += e[k]; }
#pragma unroll
            for (int off = 16; off; off >>= 1) sum += __shfl_xor(sum, off);
            float inv = 1.0f / sum;
#pragma unroll
            for (int k = 0; k < 4; ++k) att[b * SSC + lane + k * 32] = e[k] * inv;
        }
        __syncthreads();
        // ctx: thread -> (b = tid>>5, 16 consecutive d) so memory reads are contiguous
        {
            const int b     = tid >> 5;
            const int dbase = (tid & 31) * 16;
            const float* mb = memory + (size_t)b * SSC * D_DECC + dbase;
            const float* pb = att + b * SSC;
            float a16[16];
#pragma unroll
            for (int dd = 0; dd < 16; ++dd) a16[dd] = 0.0f;
            for (int s2 = 0; s2 < SSC; ++s2) {
                float p = pb[s2];
                const float* r = mb + (size_t)s2 * D_DECC;
#pragma unroll
                for (int dd = 0; dd < 16; ++dd) a16[dd] += p * r[dd];
            }
            half_t* dst = ffn + ((size_t)t * BBC + b) * 1024 + D_DECC + dbase;
#pragma unroll
            for (int dd = 0; dd < 16; ++dd) dst[dd] = (half_t)a16[dd];
        }
        __syncthreads();
    }
}

// ---------------------------------------------------------------------------
// Host launcher
// ---------------------------------------------------------------------------
extern "C" void kernel_launch(void* const* d_in, const int* in_sizes, int n_in,
                              void* d_out, int out_size, void* d_ws, size_t ws_size,
                              hipStream_t stream) {
    (void)in_sizes; (void)n_in; (void)out_size; (void)ws_size;
    const int*   inp       = (const int*)  d_in[0];
    const int*   label_i   = (const int*)  d_in[1];
    const int*   x         = (const int*)  d_in[2];
    const int*   label     = (const int*)  d_in[3];
    const float* start_emb = (const float*)d_in[4];
    const float* tok_emb   = (const float*)d_in[5];
    const float* enc_style = (const float*)d_in[6];
    const float* style_emb = (const float*)d_in[7];
    const float* Wih_f = (const float*)d_in[8];
    const float* Whh_f = (const float*)d_in[9];
    const float* bih_f = (const float*)d_in[10];
    const float* bhh_f = (const float*)d_in[11];
    const float* Wih_b = (const float*)d_in[12];
    const float* Whh_b = (const float*)d_in[13];
    const float* bih_b = (const float*)d_in[14];
    const float* bhh_b = (const float*)d_in[15];
    const float* Wih_d = (const float*)d_in[16];
    const float* Whh_d = (const float*)d_in[17];
    const float* bih_d = (const float*)d_in[18];
    const float* bhh_d = (const float*)d_in[19];
    const float* W_tr  = (const float*)d_in[20];
    const float* W_f1  = (const float*)d_in[21];
    const float* b_f1  = (const float*)d_in[22];
    const float* W_f2  = (const float*)d_in[23];

    char* base = (char*)d_ws;
    size_t off = 0;
    auto alloc = [&](size_t bytes) -> void* {
        void* p = base + off;
        off = (off + bytes + 255) & ~(size_t)255;
        return p;
    };

    half_t* wihf  = (half_t*)alloc((size_t)1024 * 128 * 2);
    half_t* whh_e = (half_t*)alloc((size_t)2 * 1024 * 256 * 2);   // [fwd|bwd]
    half_t* wihb  = (half_t*)alloc((size_t)1024 * 128 * 2);
    half_t* wihd  = (half_t*)alloc((size_t)2048 * 128 * 2);
    half_t* whhd  = (half_t*)alloc((size_t)2048 * 512 * 2);
    half_t* wtr   = (half_t*)alloc((size_t)512 * 512 * 2);
    half_t* wf1   = (half_t*)alloc((size_t)512 * 1024 * 2);
    half_t* wf2   = (half_t*)alloc((size_t)VOCABC * 512 * 2);
    half_t* embE  = (half_t*)alloc((size_t)4096 * 128 * 2);
    half_t* embD  = (half_t*)alloc((size_t)4096 * 128 * 2);
    float*  Xe    = (float*) alloc((size_t)2 * 4096 * 1024 * 4);
    float*  Xdg   = (float*) alloc((size_t)4096 * 2048 * 4);
    float*  memf  = (float*) alloc((size_t)BBC * SSC * 512 * 4);
    float*  cout_ = (float*) alloc((size_t)2 * BBC * 256 * 4);
    float*  hd    = (float*) alloc((size_t)BBC * 512 * 4);
    float*  cd    = (float*) alloc((size_t)BBC * 512 * 4);
    half_t* ffn   = (half_t*)alloc((size_t)4096 * 1024 * 2);
    half_t* hid   = (half_t*)alloc((size_t)4096 * 512 * 2);

    auto cvt = [&](const float* src, half_t* dst, int n) {
        f32_to_f16_kernel<<<(n + 255) / 256, 256, 0, stream>>>(src, dst, n);
    };
    cvt(Wih_f, wihf, 1024 * 128);
    cvt(Whh_f, whh_e, 1024 * 256);
    cvt(Whh_b, whh_e + 1024 * 256, 1024 * 256);
    cvt(Wih_b, wihb, 1024 * 128);
    cvt(Wih_d, wihd, 2048 * 128);
    cvt(Whh_d, whhd, 2048 * 512);
    cvt(W_tr,  wtr,  512 * 512);
    cvt(W_f1,  wf1,  512 * 1024);
    cvt(W_f2,  wf2,  VOCABC * 512);

    gather_enc_emb_kernel<<<(SSC * BBC * 128 + 255) / 256, 256, 0, stream>>>(inp, tok_emb, embE);
    gather_dec_emb_kernel<<<(TTC * BBC * 128 + 255) / 256, 256, 0, stream>>>(x, tok_emb, start_emb, embD);

    // X-parts of LSTM gates (biases folded): waves = (M/16)*(N/64), 8 waves/block
    wmma_gemm_kernel<<<512, 256, 0, stream>>>(embE, 128, wihf, 128, bih_f, bhh_f,
                                              Xe, (half_t*)nullptr, 1024, 4096, 1024, 128, 0);
    wmma_gemm_kernel<<<512, 256, 0, stream>>>(embE, 128, wihb, 128, bih_b, bhh_b,
                                              Xe + (size_t)4096 * 1024, (half_t*)nullptr, 1024, 4096, 1024, 128, 0);
    wmma_gemm_kernel<<<1024, 256, 0, stream>>>(embD, 128, wihd, 128, bih_d, bhh_d,
                                               Xdg, (half_t*)nullptr, 2048, 4096, 2048, 128, 0);

    // Recurrent encoder (fwd + bwd blocks), cell transform, recurrent decoder
    enc_rnn_kernel<<<2, 512, 0, stream>>>(Xe, whh_e, label_i, enc_style, memf, cout_);
    ctran_kernel<<<1, 256, 0, stream>>>(cout_, wtr, label, style_emb, cd, hd);
    dec_rnn_kernel<<<1, 1024, 0, stream>>>(Xdg, whhd, hd, cd, memf, ffn);

    // FFN1: hid = lrelu(ffn @ W_f1^T + b_f1) in f16
    wmma_gemm_kernel<<<256, 256, 0, stream>>>(ffn, 1024, wf1, 1024, b_f1, (const float*)nullptr,
                                              (float*)nullptr, hid, 512, 4096, 512, 1024, 1);

    // Vocab projection: 4096 x 32000, K=512 -> d_out[B][T][V]
    vocab_gemm_kernel<<<16000, 256, 0, stream>>>(hid, wf2, (float*)d_out);
}
